// GravnetModel_79345225826506
// MI455X (gfx1250) — matrix operand order repaired and checked by reading, not compiled, not run on hardware
//
#include <hip/hip_runtime.h>

// ---------------------------------------------------------------------------
// GravNet forward for MI455X (gfx1250).  All GEMMs run through
// v_wmma_f32_16x16x32_bf16 (fp32 accumulate).  kNN candidate tiles are staged
// memory->LDS with global_load_async_to_lds_b128 (ASYNCcnt).  Events are
// contiguous blocks of 2000 rows, so segment reductions are per-event blocks.
// ---------------------------------------------------------------------------

#define NN   16000   // B*H hits
#define BEV  8
#define HH   2000
#define KNN  40

typedef __attribute__((ext_vector_type(16))) __bf16 v16bf;
typedef __attribute__((ext_vector_type(8)))  __bf16 v8bf;
typedef __attribute__((ext_vector_type(8)))  float  v8f;

// ---------------------------- batchnorm ------------------------------------
// One block per column: biased mean/var over N rows -> scale/shift.
__global__ void bn_stats_kernel(const float* __restrict__ X, int N, int F,
                                const float* __restrict__ g,
                                const float* __restrict__ b,
                                float* __restrict__ ss) {
  const int f = blockIdx.x;
  __shared__ float s1[256], s2[256];
  float a = 0.f, c = 0.f;
  for (int r = threadIdx.x; r < N; r += 256) {
    float v = X[(size_t)r * F + f];
    a += v; c += v * v;
  }
  s1[threadIdx.x] = a; s2[threadIdx.x] = c;
  __syncthreads();
  for (int st = 128; st > 0; st >>= 1) {
    if (threadIdx.x < st) {
      s1[threadIdx.x] += s1[threadIdx.x + st];
      s2[threadIdx.x] += s2[threadIdx.x + st];
    }
    __syncthreads();
  }
  if (threadIdx.x == 0) {
    float mean = s1[0] / (float)N;
    float var  = s2[0] / (float)N - mean * mean;
    float sc   = g[f] * rsqrtf(var + 1e-5f);
    ss[f]     = sc;
    ss[F + f] = b[f] - mean * sc;
  }
}

__global__ void bn_apply_kernel(const float* __restrict__ X, float* __restrict__ Y,
                                const float* __restrict__ ss, int N, int F) {
  int idx = blockIdx.x * blockDim.x + threadIdx.x;
  if (idx >= N * F) return;
  int f = idx % F;
  Y[idx] = X[idx] * ss[f] + ss[F + f];
}

// ------------------------- global exchange ---------------------------------
// grid (BEV, F): per-event mean/min/max of one feature over 2000 rows.
__global__ void event_stats_kernel(const float* __restrict__ X, int F,
                                   float* __restrict__ ES) {
  const int ev = blockIdx.x, f = blockIdx.y;
  __shared__ float sh0[256], sh1[256], sh2[256];
  float s = 0.f, mn = 3.4e38f, mx = -3.4e38f;
  for (int r = threadIdx.x; r < HH; r += 256) {
    float v = X[(size_t)(ev * HH + r) * F + f];
    s += v; mn = fminf(mn, v); mx = fmaxf(mx, v);
  }
  sh0[threadIdx.x] = s; sh1[threadIdx.x] = mn; sh2[threadIdx.x] = mx;
  __syncthreads();
  for (int st = 128; st > 0; st >>= 1) {
    if (threadIdx.x < st) {
      sh0[threadIdx.x] += sh0[threadIdx.x + st];
      sh1[threadIdx.x] = fminf(sh1[threadIdx.x], sh1[threadIdx.x + st]);
      sh2[threadIdx.x] = fmaxf(sh2[threadIdx.x], sh2[threadIdx.x + st]);
    }
    __syncthreads();
  }
  if (threadIdx.x == 0) {
    ES[(ev * 3 + 0) * F + f] = sh0[0] * (1.0f / HH);
    ES[(ev * 3 + 1) * F + f] = sh1[0];
    ES[(ev * 3 + 2) * F + f] = sh2[0];
  }
}

// y[i] = [mean | min | max | x]  (stride 4F)
__global__ void global_exchange_kernel(const float* __restrict__ X,
                                       const float* __restrict__ ES,
                                       float* __restrict__ Y, int F, int N) {
  int idx = blockIdx.x * blockDim.x + threadIdx.x;
  if (idx >= N * F) return;
  int i = idx / F, f = idx % F;
  int ev = i / HH;
  float* yr = Y + (size_t)i * 4 * F;
  yr[f]         = ES[(ev * 3 + 0) * F + f];
  yr[F + f]     = ES[(ev * 3 + 1) * F + f];
  yr[2 * F + f] = ES[(ev * 3 + 2) * F + f];
  yr[3 * F + f] = X[(size_t)i * F + f];
}

// ------------------------- bf16 packing ------------------------------------
// A: (N x K) f32 row-major -> (N x Kpad) bf16 zero-padded.
__global__ void convert_a_kernel(const float* __restrict__ A, __bf16* __restrict__ Abf,
                                 int N, int K, int Kpad) {
  int idx = blockIdx.x * blockDim.x + threadIdx.x;
  if (idx >= N * Kpad) return;
  int i = idx / Kpad, k = idx % Kpad;
  Abf[idx] = (k < K) ? (__bf16)A[(size_t)i * K + k] : (__bf16)0.0f;
}

// W: (K x dout) f32 -> Wt: (doutPad x Kpad) bf16 (transposed, zero padded) so
// each B fragment column is 16 contiguous K values per lane (contiguous k order, same as the A fragment).
__global__ void convert_w_kernel(const float* __restrict__ W, __bf16* __restrict__ Wt,
                                 int K, int Kpad, int dout, int doutPad) {
  int idx = blockIdx.x * blockDim.x + threadIdx.x;
  if (idx >= doutPad * Kpad) return;
  int n = idx / Kpad, k = idx % Kpad;
  Wt[idx] = (k < K && n < dout) ? (__bf16)W[(size_t)k * dout + n] : (__bf16)0.0f;
}

// ---------------------------- WMMA GEMM ------------------------------------
// C(N x dout) = A(N x Kpad)bf16 @ Wt^T + bias, fused activation.
// 256 threads = 8 waves; each wave owns a 16x16 tile (4 along M, 2 along N).
// Fragment layouts (A and B share contiguous k order, not the ISA table order):
//   A 16x32 bf16: lanes<16 row=l, K = k0+0..15;          lanes>=16 K = k0+16..31
//   B 32x16 bf16: lanes<16 col=l, K = k0+0..15;          lanes>=16 K = k0+16..31
//   D 16x16 f32 : vgpr r -> row r (+8 for upper lanes), col = lane&15
__global__ void wmma_gemm_bf16(const __bf16* __restrict__ A,
                               const __bf16* __restrict__ Wt,
                               const float* __restrict__ bias,
                               float* __restrict__ C,
                               int N, int Kpad, int dout, int doutPad, int act) {
  const int lane = threadIdx.x & 31;
  const int wave = threadIdx.x >> 5;
  const int tileM = blockIdx.x * 4 + (wave & 3);
  const int tileN = blockIdx.y * 2 + (wave >> 2);
  if (tileM * 16 >= N) return;
  if (tileN * 16 >= doutPad) return;

  const int half = lane >> 4;   // 0: lower half lanes, 1: upper
  const int l16  = lane & 15;

  const __bf16* arow = A  + (size_t)(tileM * 16 + l16) * Kpad + half * 16;
  const __bf16* brow = Wt + (size_t)(tileN * 16 + l16) * Kpad + half * 16;

  v8f acc = {};
  for (int k0 = 0; k0 < Kpad; k0 += 32) {
    v8bf a_lo = *(const v8bf*)(arow + k0);
    v8bf a_hi = *(const v8bf*)(arow + k0 + 8);
    v16bf a;
#pragma unroll
    for (int i = 0; i < 8; ++i) { a[i] = a_lo[i]; a[i + 8] = a_hi[i]; }
    v16bf b = *(const v16bf*)(brow + k0);
    acc = __builtin_amdgcn_wmma_f32_16x16x32_bf16(
        /*neg_a=*/false, a, /*neg_b=*/false, b,
        /*c_mod=*/(short)0, acc, /*reuse_a=*/false, /*reuse_b=*/false);
  }

  const int col = tileN * 16 + l16;
  if (col < dout) {
    float bv = bias ? bias[col] : 0.0f;
#pragma unroll
    for (int r = 0; r < 8; ++r) {
      int row = tileM * 16 + half * 8 + r;
      float v = acc[r] + bv;
      if (act == 1)      v = tanhf(v);
      else if (act == 2) v = fmaxf(v, 0.0f);
      C[(size_t)row * dout + col] = v;
    }
  }
}

// ------------------------------- kNN ---------------------------------------
// One thread per query; candidate tiles staged memory->LDS with the CDNA5
// async DMA path (global_load_async_to_lds_b128, ASYNCcnt) — one 16-byte
// point per lane, no VGPR round-trip.  Each thread maintains a 40-entry
// replace-the-max list; w = exp(-10 d^2).
__global__ void knn_kernel(const float* __restrict__ S,
                           int* __restrict__ kidx, float* __restrict__ kw) {
  const int ev = blockIdx.x;
  const int q  = blockIdx.y * blockDim.x + threadIdx.x;
  const int base = ev * HH;
  const bool valid = (q < HH);
  __shared__ float tile[128][4];

  float s0 = 0.f, s1 = 0.f, s2 = 0.f, s3 = 0.f;
  if (valid) {
    const float* sp = S + (size_t)(base + q) * 4;
    s0 = sp[0]; s1 = sp[1]; s2 = sp[2]; s3 = sp[3];
  }
  float bd[KNN]; int bi[KNN];
#pragma unroll
  for (int k = 0; k < KNN; ++k) { bd[k] = 3.4e38f; bi[k] = base; }
  float worst = 3.4e38f; int wslot = 0;

  for (int t0 = 0; t0 < HH; t0 += 128) {
    int tc = HH - t0; if (tc > 128) tc = 128;
    __syncthreads();  // previous tile fully consumed
    if ((int)threadIdx.x < tc) {
      unsigned int ldsoff = (unsigned int)(size_t)&tile[threadIdx.x][0];
      unsigned long long gaddr =
          (unsigned long long)(size_t)(S + (size_t)(base + t0 + threadIdx.x) * 4);
      // memory -> LDS, 16B per lane, tracked by ASYNCcnt
      asm volatile("global_load_async_to_lds_b128 %0, %1, off"
                   :: "v"(ldsoff), "v"(gaddr) : "memory");
    }
    asm volatile("s_wait_asynccnt 0x0" ::: "memory");  // own wave's DMAs done
    __syncthreads();                                   // => all waves' DMAs done
    if (valid) {
      for (int i = 0; i < tc; ++i) {
        float d0 = s0 - tile[i][0], d1 = s1 - tile[i][1];
        float d2 = s2 - tile[i][2], d3 = s3 - tile[i][3];
        float dd = d0 * d0 + d1 * d1 + d2 * d2 + d3 * d3;
        if (dd < worst) {
          bd[wslot] = dd; bi[wslot] = base + t0 + i;
          worst = bd[0]; wslot = 0;
#pragma unroll
          for (int k = 1; k < KNN; ++k)
            if (bd[k] > worst) { worst = bd[k]; wslot = k; }
        }
      }
    }
  }
  if (valid) {
    for (int k = 0; k < KNN; ++k) {
      kidx[(size_t)(base + q) * KNN + k] = bi[k];
      kw  [(size_t)(base + q) * KNN + k] = __expf(-10.0f * bd[k]);
    }
  }
}

// Weighted mean/max over K neighbors of the 22 learned props; writes into the
// concat buffer [x | mean(22) | max(22)] at column offset cin. 4 rows/block.
__global__ void aggregate_kernel(const float* __restrict__ Hp,
                                 const int* __restrict__ kidx,
                                 const float* __restrict__ kw,
                                 float* __restrict__ out, int cin) {
  const int i = blockIdx.x * 4 + (threadIdx.x >> 5);
  const int j = threadIdx.x & 31;
  if (i >= NN || j >= 22) return;
  const int stride = cin + 44;
  const int*   ip = kidx + (size_t)i * KNN;
  const float* wp = kw   + (size_t)i * KNN;
  float s = 0.f, m = -3.4e38f;
  for (int k = 0; k < KNN; ++k) {
    float v = Hp[(size_t)ip[k] * 22 + j] * wp[k];
    s += v; m = fmaxf(m, v);
  }
  out[(size_t)i * stride + cin + j]      = s * (1.0f / KNN);
  out[(size_t)i * stride + cin + 22 + j] = m;
}

// Copy an (N x F) matrix into a strided destination at a column offset.
__global__ void copy_cols_kernel(const float* __restrict__ src, float* __restrict__ dst,
                                 int N, int F, int dstStride, int dstOff) {
  int idx = blockIdx.x * blockDim.x + threadIdx.x;
  if (idx >= N * F) return;
  int i = idx / F, f = idx % F;
  dst[(size_t)i * dstStride + dstOff + f] = src[idx];
}

// ---------------------------------------------------------------------------
extern "C" void kernel_launch(void* const* d_in, const int* in_sizes, int n_in,
                              void* d_out, int out_size, void* d_ws, size_t ws_size,
                              hipStream_t stream) {
  (void)in_sizes; (void)n_in; (void)out_size; (void)ws_size;
  const int N = NN;

  // --- param indexing (JAX tree order: top-level insertion order x, params, batch;
  //     nested dicts sorted by key; lists in order) ---
  // params: blocks(4x18 leaves), bn0(b,g), input(W,b), o1, o2, o3, postgn(4x4)
  auto P = [&](int i) { return (const float*)d_in[i]; };
  const float* x_in = P(0);
  auto BK = [&](int b, int o) { return (const float*)d_in[1 + b * 18 + o]; };
  // block leaf offsets: 0 bn1.b 1 bn1.g 2 bn2.b 3 bn2.g 4 bn3.b 5 bn3.g
  //                     6 fc1.W 7 fc1.b 8 fc2.W 9 fc2.b 10 lin_h.W 11 lin_h.b
  //                     12 lin_out.W 13 lin_out.b 14 lin_s.W 15 lin_s.b 16 out.W 17 out.b
  const int I_BN0B = 73, I_BN0G = 74, I_INW = 75, I_INB = 76;
  const int I_O1W = 77, I_O1B = 78, I_O2W = 79, I_O2B = 80, I_O3W = 81, I_O3B = 82;
  const int I_PG = 83; // postgn i: +0 bn.b +1 bn.g +2 lin.W +3 lin.b

  // --- workspace carve ---
  char* wsb = (char*)d_ws;
  size_t off = 0;
  auto alloc = [&](size_t bytes) -> void* {
    void* p = wsb + off;
    off += (bytes + 255) & ~(size_t)255;
    return p;
  };
  float*  B0    = (float*)alloc((size_t)N * 96 * 4);   // h
  float*  B1    = (float*)alloc((size_t)N * 384 * 4);  // concat / exchange
  float*  B2    = (float*)alloc((size_t)N * 128 * 4);  // gemm out
  float*  OUTS  = (float*)alloc((size_t)N * 384 * 4);
  __bf16* ABF   = (__bf16*)alloc((size_t)N * 384 * 2);
  __bf16* WBF   = (__bf16*)alloc((size_t)128 * 384 * 2);
  float*  SS    = (float*)alloc(512 * 4);
  float*  ES    = (float*)alloc((size_t)BEV * 3 * 96 * 4);
  float*  SBUF  = (float*)alloc((size_t)N * 4 * 4);
  float*  HPROP = (float*)alloc((size_t)N * 22 * 4);
  int*    KIDX  = (int*)alloc((size_t)N * KNN * 4);
  float*  KW    = (float*)alloc((size_t)N * KNN * 4);

  auto up32 = [](int k) { return (k + 31) & ~31; };
  auto up16 = [](int k) { return (k + 15) & ~15; };

  // Pack activations to bf16 once per distinct input matrix.
  auto convA = [&](const float* src, int K) -> int {
    int Kpad = up32(K);
    int t = N * Kpad;
    convert_a_kernel<<<(t + 255) / 256, 256, 0, stream>>>(src, ABF, N, K, Kpad);
    return Kpad;
  };
  // WMMA GEMM on pre-packed A.
  auto gemmP = [&](int K, int Kpad, const float* W, const float* bias,
                   int dout, float* Cout, int act) {
    int doutPad = up16(dout);
    int tw = doutPad * Kpad;
    convert_w_kernel<<<(tw + 255) / 256, 256, 0, stream>>>(W, WBF, K, Kpad, dout, doutPad);
    dim3 g((N + 63) / 64, (doutPad + 31) / 32);
    wmma_gemm_bf16<<<g, 256, 0, stream>>>(ABF, WBF, bias, Cout, N, Kpad, dout, doutPad, act);
  };
  auto gemm = [&](const float* Af32, int K, const float* W, const float* bias,
                  int dout, float* Cout, int act) {
    int Kpad = convA(Af32, K);
    gemmP(K, Kpad, W, bias, dout, Cout, act);
  };
  auto bn = [&](const float* X, int F, const float* g_, const float* b_, float* Y) {
    bn_stats_kernel<<<F, 256, 0, stream>>>(X, N, F, g_, b_, SS);
    int t = N * F;
    bn_apply_kernel<<<(t + 255) / 256, 256, 0, stream>>>(X, Y, SS, N, F);
  };
  auto gexch = [&](const float* X, int F, float* Y) {
    event_stats_kernel<<<dim3(BEV, F), 256, 0, stream>>>(X, F, ES);
    int t = N * F;
    global_exchange_kernel<<<(t + 255) / 256, 256, 0, stream>>>(X, ES, Y, F, N);
  };
  auto ccopy = [&](const float* src, float* dst, int F, int stride, int o) {
    int t = N * F;
    copy_cols_kernel<<<(t + 255) / 256, 256, 0, stream>>>(src, dst, N, F, stride, o);
  };

  // ---------------- stage 0: bn0 + global_exchange + input linear ----------
  bn(x_in, 8, P(I_BN0G), P(I_BN0B), B0);
  gexch(B0, 8, B1);                                   // (N,32)
  gemm(B1, 32, P(I_INW), P(I_INB), 64, B0, 0);        // h = (N,64)

  // ---------------- gravnet blocks -----------------------------------------
  int cin = 64;
  const float* h = B0;
  for (int b = 0; b < 4; ++b) {
    int Kpad = convA(h, cin);                         // pack h once for s & h
    gemmP(cin, Kpad, BK(b, 14), BK(b, 15), 4,  SBUF,  0); // lin_s -> (N,4)
    gemmP(cin, Kpad, BK(b, 10), BK(b, 11), 22, HPROP, 0); // lin_h -> (N,22)
    knn_kernel<<<dim3(BEV, (HH + 127) / 128), 128, 0, stream>>>(SBUF, KIDX, KW);
    ccopy(h, B1, cin, cin + 44, 0);                   // concat: x part
    aggregate_kernel<<<N / 4, 128, 0, stream>>>(HPROP, KIDX, KW, B1, cin);
    gemm(B1, cin + 44, BK(b, 12), BK(b, 13), 96, B2, 0);      // lin_out
    bn(B2, 96, BK(b, 1), BK(b, 0), B1);                       // bn1
    gemm(B1, 96, BK(b, 6), BK(b, 7), 128, B2, 1);             // fc1 + tanh
    bn(B2, 128, BK(b, 3), BK(b, 2), B1);                      // bn2
    gemm(B1, 128, BK(b, 8), BK(b, 9), 96, B2, 1);             // fc2 + tanh
    gexch(B2, 96, B1);                                        // (N,384)
    gemm(B1, 384, BK(b, 16), BK(b, 17), 96, B2, 1);           // out + tanh
    bn(B2, 96, BK(b, 5), BK(b, 4), B0);                       // bn3 -> new h
    ccopy(B0, OUTS, 96, 384, b * 96);
    h = B0; cin = 96;
  }

  // ---------------- post-gravnet dense stack --------------------------------
  const float* hp = OUTS;
  int din = 384;
  for (int i = 0; i < 4; ++i) {
    gemm(hp, din, P(I_PG + i * 4 + 2), P(I_PG + i * 4 + 3), 128, B1, 2); // relu
    bn(B1, 128, P(I_PG + i * 4 + 1), P(I_PG + i * 4 + 0), B2);
    hp = B2; din = 128;
  }
  gemm(B2, 128, P(I_O1W), P(I_O1B), 64, B1, 2);
  gemm(B1, 64,  P(I_O2W), P(I_O2B), 64, B2, 2);
  gemm(B2, 64,  P(I_O3W), P(I_O3B), 8, (float*)d_out, 0);
}
